// CogAgentDecoderLayer_49855980372286
// MI455X (gfx1250) — matrix operand order repaired and checked by reading, not compiled, not run on hardware
//
#include <hip/hip_runtime.h>
#include <math.h>

typedef __attribute__((ext_vector_type(16))) __bf16 v16bf;
typedef __attribute__((ext_vector_type(8)))  __bf16 v8bf;
typedef __attribute__((ext_vector_type(8)))  float  v8f;

static constexpr int B_  = 2;
static constexpr int L_  = 2048;
static constexpr int H_  = 2048;
static constexpr int NH_ = 16;
static constexpr int HD_ = 128;
static constexpr int CH_ = 1024;
static constexpr int CC_ = 1024;
static constexpr int CHD_ = 64;
static constexpr int I_  = 5504;
static constexpr int LE_ = 1024;
static constexpr int BL_  = B_ * L_;    // 4096 tokens
static constexpr int BLE_ = B_ * LE_;   // 2048 encoder tokens

// ---------------- WMMA fragment helpers (CDNA5 ISA 7.12.2) ----------------
// A (16x32 bf16): lane half 0 holds K 0..15; half 1 holds K 16..31 (contiguous)
// B (32x16 bf16): lane half 0 holds K 0..15; half 1 holds K 16..31 (contiguous)
__device__ __forceinline__ v8f wmma_bf16(v16bf a, v16bf b, v8f c) {
    return __builtin_amdgcn_wmma_f32_16x16x32_bf16(false, a, false, b, (short)0, c,
                                                   false, false);
}

// A-fragment: two 16B vector loads. p = row base + k0 (16B-aligned by construction).
__device__ __forceinline__ v16bf load_a_frag(const __bf16* p, int half) {
    v8bf lo = *(const v8bf*)(p + half * 16);       // K = half*16 .. half*16+7
    v8bf hi = *(const v8bf*)(p + half * 16 + 8);   // K = half*16+8 .. +7
    return __builtin_shufflevector(lo, hi, 0, 1, 2, 3, 4, 5, 6, 7,
                                   8, 9, 10, 11, 12, 13, 14, 15);
}

// bf16 B-fragment with K contiguous in memory: two 16B vector loads.
__device__ __forceinline__ v16bf load_b_frag_contig(const __bf16* p, int half) {
    v8bf lo = *(const v8bf*)(p + half * 16);
    v8bf hi = *(const v8bf*)(p + half * 16 + 8);
    return __builtin_shufflevector(lo, hi, 0, 1, 2, 3, 4, 5, 6, 7,
                                   8, 9, 10, 11, 12, 13, 14, 15);
}

// f32 weight B-fragment: p = W + (k0 + half*16)*N + col. Compile-time N makes
// every element a constant immediate offset from one base -> independent,
// clause-able global_load_b32, no address-add chain.
template <int N>
__device__ __forceinline__ v16bf load_w_frag(const float* __restrict__ p) {
    v16bf b;
#pragma unroll
    for (int e = 0; e < 16; ++e) b[e] = (__bf16)p[(size_t)e * N];
    return b;
}

// ---------------- vision-expert mask ----------------
__global__ void k_mask(const int* __restrict__ tt, unsigned char* __restrict__ vm) {
    int idx = blockIdx.x * 256 + threadIdx.x;
    if (idx >= BL_) return;
    int l = idx % L_;
    int m = 0;
    if (l + 1 < L_) m = (tt[idx] == 1 && tt[idx + 1] == 1) ? 1 : 0;
    vm[idx] = (unsigned char)m;
}

// ---------------- RMSNorm (f32 in, bf16 out), one block per token ----------------
__global__ void k_rmsnorm(const float* __restrict__ x, const float* __restrict__ w,
                          __bf16* __restrict__ out) {
    int t = blockIdx.x;
    const float* xr = x + (size_t)t * H_;
    __shared__ float red[256];
    float s = 0.f;
    for (int i = threadIdx.x; i < H_; i += 256) { float v = xr[i]; s += v * v; }
    red[threadIdx.x] = s;
    __syncthreads();
    for (int st = 128; st > 0; st >>= 1) {
        if (threadIdx.x < st) red[threadIdx.x] += red[threadIdx.x + st];
        __syncthreads();
    }
    float inv = rsqrtf(red[0] / (float)H_ + 1e-6f);
    for (int i = threadIdx.x; i < H_; i += 256)
        out[(size_t)t * H_ + i] = (__bf16)(w[i] * xr[i] * inv);
}

// ---------------- f32 -> bf16 ----------------
__global__ void k_cvt(const float* __restrict__ in, __bf16* __restrict__ out, int n) {
    int idx = blockIdx.x * 256 + threadIdx.x;
    if (idx < n) out[idx] = (__bf16)in[idx];
}

// ---------------- RoPE in place on q,k sections of mixed [BL, 3H] ----------------
__global__ void k_rope(__bf16* __restrict__ mixed, const int* __restrict__ pos) {
    int idx = blockIdx.x * 256 + threadIdx.x;
    const int total = BL_ * NH_ * (HD_ / 2);
    if (idx >= total) return;
    int d = idx & 63;            // 0..HD/2-1
    int r = idx >> 6;
    int hh = r % NH_; r /= NH_;
    int l = r % L_;  int b = r / L_;
    float p = (float)pos[l];
    float inv = __expf(-((2.f * d) / (float)HD_) * 9.210340371976184f); // ln(1e4)
    float ang = p * inv;
    float c = __cosf(ang), s = __sinf(ang);
    size_t base = (size_t)(b * L_ + l) * (3 * H_);
    size_t qo = base + (size_t)hh * HD_ + d;
    size_t ko = base + H_ + (size_t)hh * HD_ + d;
    float q1 = (float)mixed[qo], q2 = (float)mixed[qo + HD_ / 2];
    mixed[qo]            = (__bf16)(q1 * c - q2 * s);
    mixed[qo + HD_ / 2]  = (__bf16)(q2 * c + q1 * s);
    float k1 = (float)mixed[ko], k2 = (float)mixed[ko + HD_ / 2];
    mixed[ko]            = (__bf16)(k1 * c - k2 * s);
    mixed[ko + HD_ / 2]  = (__bf16)(k2 * c + k1 * s);
}

// ---------------- V transpose: token-major src -> vt[b,h,d,l] ----------------
__global__ void k_transpose_v(const __bf16* __restrict__ src, int src_stride, int src_off,
                              __bf16* __restrict__ vt, int Lk, int hd, int total) {
    int idx = blockIdx.x * 256 + threadIdx.x;
    if (idx >= total) return;
    int d = idx % hd; int r = idx / hd;
    int l = r % Lk;  r /= Lk;
    int h = r % NH_; int b = r / NH_;
    vt[((size_t)(b * NH_ + h) * hd + d) * Lk + l] =
        src[(size_t)(b * Lk + l) * src_stride + src_off + h * hd + d];
}

// ---------------- Generic WMMA GEMM ----------------
// Each wave computes MT row-tiles x one 16-col tile; the strided f32 weight
// fragment loads are amortized over MT WMMAs. N and K are compile-time so all
// weight-element offsets are instruction immediates.
// Optional dual-A / dual-B, per-row expert blend, SwiGLU combine, residual add.
template <int MT, int N, int K, bool DUALA, bool DUALB, bool MASKB, bool SWIGLU,
          bool RESADD>
__global__ void k_gemm(const __bf16* __restrict__ A, const __bf16* __restrict__ A2,
                       const float* __restrict__ B1, const float* __restrict__ B2,
                       const unsigned char* __restrict__ vm,
                       const float* __restrict__ rin, float* __restrict__ rout,
                       __bf16* __restrict__ obf) {
    int wave = threadIdx.x >> 5, lane = threadIdx.x & 31;
    int half = lane >> 4, l16 = lane & 15;
    int row0 = blockIdx.x * (16 * MT);
    int col0 = (blockIdx.y * 8 + wave) * 16;
    int bcol = col0 + l16;
    v8f acc1[MT], acc2[MT];
#pragma unroll
    for (int t = 0; t < MT; ++t) { acc1[t] = (v8f){}; acc2[t] = (v8f){}; }

    const float* wp1 = B1 + (size_t)(half * 16) * N + bcol;
    const float* wp2 = DUALB ? (B2 + (size_t)(half * 16) * N + bcol) : nullptr;

    for (int k0 = 0; k0 < K; k0 += 32) {
        v16bf af[MT], af2[MT];
#pragma unroll
        for (int t = 0; t < MT; ++t) {
            int arow = row0 + t * 16 + l16;
            af[t] = load_a_frag(A + (size_t)arow * K + k0, half);
            if constexpr (DUALA)
                af2[t] = load_a_frag(A2 + (size_t)arow * K + k0, half);
        }
        v16bf bf1 = load_w_frag<N>(wp1);
        wp1 += (size_t)32 * N;
#pragma unroll
        for (int t = 0; t < MT; ++t) acc1[t] = wmma_bf16(af[t], bf1, acc1[t]);
        if constexpr (DUALB) {
            v16bf bf2 = load_w_frag<N>(wp2);
            wp2 += (size_t)32 * N;
#pragma unroll
            for (int t = 0; t < MT; ++t) {
                if constexpr (DUALA) acc2[t] = wmma_bf16(af2[t], bf2, acc2[t]);
                else                 acc2[t] = wmma_bf16(af[t],  bf2, acc2[t]);
            }
        }
    }
#pragma unroll
    for (int t = 0; t < MT; ++t)
#pragma unroll
        for (int r = 0; r < 8; ++r) {
            int row = row0 + t * 16 + r + 8 * half;
            int col = col0 + l16;
            float v;
            if constexpr (SWIGLU) {
                float g = acc1[t][r], u = acc2[t][r];
                v = (g / (1.f + __expf(-g))) * u;
            } else if constexpr (MASKB) {
                v = vm[row] ? acc1[t][r] : acc2[t][r];
            } else {
                v = acc1[t][r];
            }
            if constexpr (RESADD)
                rout[(size_t)row * N + col] = rin[(size_t)row * N + col] + v;
            else
                obf[(size_t)row * N + col] = (__bf16)v;
        }
}

// ---------------- Flash attention: one wave = 16 query rows, key step 32 ----------------
template <int HDIM, bool CAUSAL>
__global__ void k_flash(const __bf16* __restrict__ qb, int q_stride, int q_off,
                        const __bf16* __restrict__ kb, int k_stride, int k_off,
                        const __bf16* __restrict__ vt,
                        __bf16* __restrict__ outb, int out_stride,
                        int Lq, int Lk, float scale) {
    constexpr int NC = HDIM / 32;   // QK^T K-chunks
    constexpr int OC = HDIM / 16;   // output d-chunks
    __shared__ __bf16 sh[8][16][32];
    int wave = threadIdx.x >> 5, lane = threadIdx.x & 31;
    int half = lane >> 4, l16 = lane & 15;
    int h = blockIdx.y, b = blockIdx.z;
    int q0 = (blockIdx.x * 8 + wave) * 16;

    const __bf16* qrowp = qb + (size_t)(b * Lq) * q_stride + q_off + h * HDIM;
    const __bf16* krowp = kb + (size_t)(b * Lk) * k_stride + k_off + h * HDIM;
    const __bf16* vtp   = vt + (size_t)(b * NH_ + h) * HDIM * Lk;

    v16bf qf[NC];
    int qrow = q0 + l16;
#pragma unroll
    for (int c = 0; c < NC; ++c)
        qf[c] = load_a_frag(qrowp + (size_t)qrow * q_stride + c * 32, half);

    v8f o[OC];
#pragma unroll
    for (int c = 0; c < OC; ++c) o[c] = (v8f){};
    float mr[8], lr[8];
#pragma unroll
    for (int r = 0; r < 8; ++r) { mr[r] = -3.0e38f; lr[r] = 0.f; }

    int nkt = CAUSAL ? ((q0 + 15) / 32 + 1) : (Lk / 32);
    if (nkt > Lk / 32) nkt = Lk / 32;

    for (int kt = 0; kt < nkt; ++kt) {
        v8f s0 = {}; v8f s1 = {};
#pragma unroll
        for (int c = 0; c < NC; ++c) {
            // K rows are d-contiguous: vector fragment loads
            v16bf b0 = load_b_frag_contig(
                krowp + (size_t)(kt * 32 + l16) * k_stride + c * 32, half);
            v16bf b1 = load_b_frag_contig(
                krowp + (size_t)(kt * 32 + 16 + l16) * k_stride + c * 32, half);
            s0 = wmma_bf16(qf[c], b0, s0);
            s1 = wmma_bf16(qf[c], b1, s1);
        }
        int k0g = kt * 32 + l16;
        float al[8];
#pragma unroll
        for (int r = 0; r < 8; ++r) {
            float a0 = s0[r] * scale, a1 = s1[r] * scale;
            if (CAUSAL) {
                int qr = q0 + r + 8 * half;
                if (k0g > qr)      a0 = -3.0e38f;
                if (k0g + 16 > qr) a1 = -3.0e38f;
            }
            // row max over 32 keys (16 lanes x 2 tiles)
            float t = fmaxf(a0, a1);
            for (int m = 8; m >= 1; m >>= 1) t = fmaxf(t, __shfl_xor(t, m, 16));
            float m2 = fmaxf(mr[r], t);
            al[r] = __expf(mr[r] - m2);
            mr[r] = m2;
            float p0 = __expf(a0 - m2), p1 = __expf(a1 - m2);
            s0[r] = p0; s1[r] = p1;
            float rs = p0 + p1;
            for (int m = 8; m >= 1; m >>= 1) rs += __shfl_xor(rs, m, 16);
            lr[r] = lr[r] * al[r] + rs;
        }
#pragma unroll
        for (int c = 0; c < OC; ++c)
#pragma unroll
            for (int r = 0; r < 8; ++r) o[c][r] *= al[r];
        // C-layout probs -> LDS -> A-fragment (per-wave region, same-wave DS is in-order)
#pragma unroll
        for (int r = 0; r < 8; ++r) {
            sh[wave][r + 8 * half][l16]      = (__bf16)s0[r];
            sh[wave][r + 8 * half][l16 + 16] = (__bf16)s1[r];
        }
        v16bf pf = load_a_frag(&sh[wave][l16][0], half);
        // P(16x32) x V(32xHDIM), V transposed so keys are contiguous
#pragma unroll
        for (int c = 0; c < OC; ++c) {
            v16bf bv = load_b_frag_contig(
                vtp + (size_t)(c * 16 + l16) * Lk + kt * 32, half);
            o[c] = wmma_bf16(pf, bv, o[c]);
        }
    }
#pragma unroll
    for (int c = 0; c < OC; ++c)
#pragma unroll
        for (int r = 0; r < 8; ++r) {
            int tok = q0 + r + 8 * half;
            float val = o[c][r] / lr[r];
            outb[(size_t)(b * Lq + tok) * out_stride + h * HDIM + c * 16 + l16] = (__bf16)val;
        }
}

// ---------------- host orchestration ----------------
static inline size_t al256(size_t x) { return (x + 255) & ~(size_t)255; }

extern "C" void kernel_launch(void* const* d_in, const int* in_sizes, int n_in,
                              void* d_out, int out_size, void* d_ws, size_t ws_size,
                              hipStream_t stream) {
    (void)in_sizes; (void)n_in; (void)out_size; (void)ws_size;
    const float* hs   = (const float*)d_in[0];
    const float* enc  = (const float*)d_in[1];
    const int*   tt   = (const int*)d_in[2];
    const int*   pos  = (const int*)d_in[3];
    const float* ln_in_w         = (const float*)d_in[4];
    const float* ln_post_attn_w  = (const float*)d_in[5];
    const float* ln_post_cross_w = (const float*)d_in[6];
    const float* v_qkv   = (const float*)d_in[7];
    const float* l_qkv   = (const float*)d_in[8];
    const float* v_dense = (const float*)d_in[9];
    const float* l_dense = (const float*)d_in[10];
    const float* cq_w    = (const float*)d_in[11];
    const float* ckv_w   = (const float*)d_in[12];
    const float* cdense_w = (const float*)d_in[13];
    const float* v_gate = (const float*)d_in[14];
    const float* v_up   = (const float*)d_in[15];
    const float* v_down = (const float*)d_in[16];
    const float* l_gate = (const float*)d_in[17];
    const float* l_up   = (const float*)d_in[18];
    const float* l_down = (const float*)d_in[19];

    char* base = (char*)d_ws;
    size_t off = 0;
    unsigned char* vm = (unsigned char*)(base + off); off += al256(BL_);
    float* hid = (float*)(base + off);                off += al256((size_t)BL_ * H_ * 4);
    size_t S = off; // phase-local scratch origin (reused between phases)

    k_mask<<<(BL_ + 255) / 256, 256, 0, stream>>>(tt, vm);

    // ================= Phase 1: self attention =================
    off = S;
    __bf16* xn    = (__bf16*)(base + off); off += al256((size_t)BL_ * H_ * 2);
    __bf16* mixed = (__bf16*)(base + off); off += al256((size_t)BL_ * 3 * H_ * 2);
    __bf16* vt    = (__bf16*)(base + off); off += al256((size_t)BL_ * H_ * 2);
    __bf16* ctx   = (__bf16*)(base + off); off += al256((size_t)BL_ * H_ * 2);

    k_rmsnorm<<<BL_, 256, 0, stream>>>(hs, ln_in_w, xn);
    // mixed = xn @ (vm ? v_qkv : l_qkv)   [BL, 3H]
    k_gemm<4, 3 * H_, H_, false, true, true, false, false>
        <<<dim3(BL_ / 64, (3 * H_) / 128), 256, 0, stream>>>(
            xn, nullptr, v_qkv, l_qkv, vm, nullptr, nullptr, mixed);
    k_rope<<<(BL_ * NH_ * (HD_ / 2) + 255) / 256, 256, 0, stream>>>(mixed, pos);
    k_transpose_v<<<(B_ * NH_ * L_ * HD_ + 255) / 256, 256, 0, stream>>>(
        mixed, 3 * H_, 2 * H_, vt, L_, HD_, B_ * NH_ * L_ * HD_);
    k_flash<128, true><<<dim3(L_ / 128, NH_, B_), 256, 0, stream>>>(
        mixed, 3 * H_, 0, mixed, 3 * H_, H_, vt, ctx, H_, L_, L_, 0.08838834764831845f);
    // hid = hs + ctx @ (vm ? v_dense : l_dense)
    k_gemm<4, H_, H_, false, true, true, false, true>
        <<<dim3(BL_ / 64, H_ / 128), 256, 0, stream>>>(
            ctx, nullptr, v_dense, l_dense, vm, hs, hid, nullptr);

    // ================= Phase 2: cross attention =================
    off = S;
    __bf16* ci   = (__bf16*)(base + off); off += al256((size_t)BL_ * H_ * 2);
    __bf16* cqh  = (__bf16*)(base + off); off += al256((size_t)BL_ * CC_ * 2);
    __bf16* encb = (__bf16*)(base + off); off += al256((size_t)BLE_ * CH_ * 2);
    __bf16* ckvm = (__bf16*)(base + off); off += al256((size_t)BLE_ * 2 * CC_ * 2);
    __bf16* cvt  = (__bf16*)(base + off); off += al256((size_t)BLE_ * CC_ * 2);
    __bf16* cctx = (__bf16*)(base + off); off += al256((size_t)BL_ * CC_ * 2);

    k_rmsnorm<<<BL_, 256, 0, stream>>>(hid, ln_post_cross_w, ci);
    k_gemm<4, CC_, H_, false, false, false, false, false>
        <<<dim3(BL_ / 64, CC_ / 128), 256, 0, stream>>>(
            ci, nullptr, cq_w, nullptr, nullptr, nullptr, nullptr, cqh);
    k_cvt<<<(BLE_ * CH_ + 255) / 256, 256, 0, stream>>>(enc, encb, BLE_ * CH_);
    k_gemm<4, 2 * CC_, CH_, false, false, false, false, false>
        <<<dim3(BLE_ / 64, (2 * CC_) / 128), 256, 0, stream>>>(
            encb, nullptr, ckv_w, nullptr, nullptr, nullptr, nullptr, ckvm);
    k_transpose_v<<<(B_ * NH_ * LE_ * CHD_ + 255) / 256, 256, 0, stream>>>(
        ckvm, 2 * CC_, CC_, cvt, LE_, CHD_, B_ * NH_ * LE_ * CHD_);
    k_flash<64, false><<<dim3(L_ / 128, NH_, B_), 256, 0, stream>>>(
        cqh, CC_, 0, ckvm, 2 * CC_, 0, cvt, cctx, CC_, L_, LE_, 0.125f);
    // hid = hid + cctx @ cdense_w
    k_gemm<4, H_, CC_, false, false, false, false, true>
        <<<dim3(BL_ / 64, H_ / 128), 256, 0, stream>>>(
            cctx, nullptr, cdense_w, nullptr, nullptr, hid, hid, nullptr);

    // ================= Phase 3: dual-expert SwiGLU MLP =================
    off = S;
    __bf16* mi = (__bf16*)(base + off); off += al256((size_t)BL_ * H_ * 2);
    __bf16* hv = (__bf16*)(base + off); off += al256((size_t)BL_ * I_ * 2);
    __bf16* hl = (__bf16*)(base + off); off += al256((size_t)BL_ * I_ * 2);

    k_rmsnorm<<<BL_, 256, 0, stream>>>(hid, ln_post_attn_w, mi);
    // hv = silu(mi @ v_gate) * (mi @ v_up)   (fused SwiGLU epilogue, dual-B)
    k_gemm<4, I_, H_, false, true, false, true, false>
        <<<dim3(BL_ / 64, I_ / 128), 256, 0, stream>>>(
            mi, nullptr, v_gate, v_up, nullptr, nullptr, nullptr, hv);
    k_gemm<4, I_, H_, false, true, false, true, false>
        <<<dim3(BL_ / 64, I_ / 128), 256, 0, stream>>>(
            mi, nullptr, l_gate, l_up, nullptr, nullptr, nullptr, hl);
    // d_out = hid + (vm ? hv @ v_down : hl @ l_down)   (dual-A + dual-B, row blend)
    k_gemm<2, H_, I_, true, true, true, false, true>
        <<<dim3(BL_ / 32, H_ / 128), 256, 0, stream>>>(
            hv, hl, v_down, l_down, vm, hid, (float*)d_out, nullptr);
}
